// TAttention_584115552313
// MI455X (gfx1250) — matrix operand order repaired and checked by reading, not compiled, not run on hardware
//
#include <hip/hip_runtime.h>
#include <cstdint>

// ---------------------------------------------------------------------------
// Types for CDNA5 (gfx1250) WMMA: v_wmma_f32_16x16x32_bf16, wave32.
// ---------------------------------------------------------------------------
typedef __bf16 bf16_t;
typedef bf16_t v16bf __attribute__((ext_vector_type(16)));
typedef float  v8f   __attribute__((ext_vector_type(8)));

#define D_MODEL 1024
#define SEQ     2048
#define BATCH   4
#define NHEAD   16
#define HEADD   64
#define ROWS    (BATCH * SEQ)   // 8192

// A/B fragment: 16 bf16 per lane = 32 bytes = two 16B chunks.
union ABFrag { v16bf v; uint4 u[2]; };

__device__ __forceinline__ v8f vzero8() {
  v8f z;
#pragma unroll
  for (int i = 0; i < 8; ++i) z[i] = 0.0f;
  return z;
}

// A-matrix 16x32 (MxK) bf16 layout: lanes 0-15: M=lane, K = {0..7, 16..23};
// lanes 16-31: M=lane-16, K = {8..15, 24..31}.   (ISA 7.12.2)
__device__ __forceinline__ ABFrag load_afrag(const bf16_t* rowptr, int lane, int kbase) {
  ABFrag f;
  const int c0 = kbase + ((lane & 16) ? 8 : 0);
  f.u[0] = *(const uint4*)(rowptr + c0);
  f.u[1] = *(const uint4*)(rowptr + c0 + 16);
  return f;
}

// B-matrix 32x16 (KxN) bf16 layout: lane holds one column (N = lane&15),
// lanes 0-15: K = {0..7, 16..23}, lanes 16-31: K = {8..15, 24..31}.
__device__ __forceinline__ ABFrag load_bfrag(const bf16_t* colptr, int lane, int kbase) {
  ABFrag f;
  const int c0 = kbase + ((lane & 16) ? 8 : 0);
  f.u[0] = *(const uint4*)(colptr + c0);
  f.u[1] = *(const uint4*)(colptr + c0 + 16);
  return f;
}

__device__ __forceinline__ v8f wmma_bf16(const ABFrag& a, const ABFrag& b, v8f c) {
  return __builtin_amdgcn_wmma_f32_16x16x32_bf16(false, a.v, false, b.v,
                                                 (short)0, c, false, false);
}

// ---------------------------------------------------------------------------
// CDNA5 async global->LDS copy (ASYNCcnt-tracked DMA, bypasses VGPRs).
// LDS byte address = low 32 bits of the flat address (ISA 10.2 aperture rule).
// ---------------------------------------------------------------------------
__device__ __forceinline__ void async_copy_b128(void* lds_ptr, const void* gptr) {
  const unsigned lds_addr = (unsigned)(uintptr_t)lds_ptr;
  const unsigned long long ga = (unsigned long long)(uintptr_t)gptr;
  asm volatile("global_load_async_to_lds_b128 %0, %1, off"
               :: "v"(lds_addr), "v"(ga) : "memory");
}
#define ASYNC_WAIT_8() asm volatile("s_wait_asynccnt 0x8" ::: "memory")
#define ASYNC_WAIT_0() asm volatile("s_wait_asynccnt 0x0" ::: "memory")

// Stage a 128x64 bf16 tile (row stride ldK in global) into LDS, 4 x b128/lane.
__device__ __forceinline__ void stage_tile_async(bf16_t* ldsbase, const bf16_t* g,
                                                 int ldK, int tid) {
#pragma unroll
  for (int i = 0; i < 4; ++i) {
    const int c   = tid + i * 256;       // 0..1023 chunks of 8 bf16
    const int row = c >> 3;
    const int col = (c & 7) * 8;
    async_copy_b128(ldsbase + row * 64 + col, g + (size_t)row * ldK + col);
  }
}

// ---------------------------------------------------------------------------
// fp32 -> bf16 conversion (weights), 4 elems/thread.
// ---------------------------------------------------------------------------
__global__ __launch_bounds__(256, 1)
void cvt_bf16_kernel(const float* __restrict__ in, bf16_t* __restrict__ out, int n) {
  const int i = (blockIdx.x * 256 + threadIdx.x) * 4;
  if (i >= n) return;
  const float4 f = *(const float4*)(in + i);
  union { bf16_t e[4]; uint2 u; } pk;
  pk.e[0] = (bf16_t)f.x; pk.e[1] = (bf16_t)f.y;
  pk.e[2] = (bf16_t)f.z; pk.e[3] = (bf16_t)f.w;
  *(uint2*)(out + i) = pk.u;
}

// ---------------------------------------------------------------------------
// LayerNorm (fp32 in -> bf16 out). One block per row of 1024.
// ---------------------------------------------------------------------------
__global__ __launch_bounds__(256, 1)
void ln_bf16_kernel(const float* __restrict__ x, const float* __restrict__ g,
                    const float* __restrict__ be, bf16_t* __restrict__ y) {
  const int row = blockIdx.x, tid = threadIdx.x;
  const int lane = tid & 31, wave = tid >> 5;
  const float4 xv = *(const float4*)(x + (size_t)row * D_MODEL + tid * 4);
  float s  = xv.x + xv.y + xv.z + xv.w;
  float s2 = xv.x * xv.x + xv.y * xv.y + xv.z * xv.z + xv.w * xv.w;
#pragma unroll
  for (int o = 16; o > 0; o >>= 1) {
    s  += __shfl_xor(s,  o, 32);
    s2 += __shfl_xor(s2, o, 32);
  }
  __shared__ float sh1[8], sh2[8];
  if (lane == 0) { sh1[wave] = s; sh2[wave] = s2; }
  __syncthreads();
  float ts = 0.f, ts2 = 0.f;
#pragma unroll
  for (int i = 0; i < 8; ++i) { ts += sh1[i]; ts2 += sh2[i]; }
  const float mu  = ts * (1.0f / D_MODEL);
  const float var = ts2 * (1.0f / D_MODEL) - mu * mu;
  const float rs  = rsqrtf(var + 1e-5f);
  union { bf16_t e[4]; uint2 u; } pk;
  const float xx[4] = {xv.x, xv.y, xv.z, xv.w};
#pragma unroll
  for (int i = 0; i < 4; ++i) {
    const int c = tid * 4 + i;
    pk.e[i] = (bf16_t)((xx[i] - mu) * rs * g[c] + be[c]);
  }
  *(uint2*)(y + (size_t)row * D_MODEL + tid * 4) = pk.u;
}

// ---------------------------------------------------------------------------
// xt = x + att (fp32 out) ; y = bf16(LN(xt)).
// ---------------------------------------------------------------------------
__global__ __launch_bounds__(256, 1)
void add_ln_kernel(const float* __restrict__ x, const float* __restrict__ att,
                   const float* __restrict__ g, const float* __restrict__ be,
                   float* __restrict__ xt, bf16_t* __restrict__ y) {
  const int row = blockIdx.x, tid = threadIdx.x;
  const int lane = tid & 31, wave = tid >> 5;
  const size_t base = (size_t)row * D_MODEL + tid * 4;
  const float4 xv = *(const float4*)(x + base);
  const float4 av = *(const float4*)(att + base);
  float4 xa;
  xa.x = xv.x + av.x; xa.y = xv.y + av.y;
  xa.z = xv.z + av.z; xa.w = xv.w + av.w;
  *(float4*)(xt + base) = xa;
  float s  = xa.x + xa.y + xa.z + xa.w;
  float s2 = xa.x * xa.x + xa.y * xa.y + xa.z * xa.z + xa.w * xa.w;
#pragma unroll
  for (int o = 16; o > 0; o >>= 1) {
    s  += __shfl_xor(s,  o, 32);
    s2 += __shfl_xor(s2, o, 32);
  }
  __shared__ float sh1[8], sh2[8];
  if (lane == 0) { sh1[wave] = s; sh2[wave] = s2; }
  __syncthreads();
  float ts = 0.f, ts2 = 0.f;
#pragma unroll
  for (int i = 0; i < 8; ++i) { ts += sh1[i]; ts2 += sh2[i]; }
  const float mu  = ts * (1.0f / D_MODEL);
  const float var = ts2 * (1.0f / D_MODEL) - mu * mu;
  const float rs  = rsqrtf(var + 1e-5f);
  union { bf16_t e[4]; uint2 u; } pk;
  const float xx[4] = {xa.x, xa.y, xa.z, xa.w};
#pragma unroll
  for (int i = 0; i < 4; ++i) {
    const int c = tid * 4 + i;
    pk.e[i] = (bf16_t)((xx[i] - mu) * rs * g[c] + be[c]);
  }
  *(uint2*)(y + base) = pk.u;
}

// ---------------------------------------------------------------------------
// WMMA GEMM: C[M,N] = A[M,K] * Bw[N,K]^T  (+bias, relu, residual).
// Block = 256 thr = 8 waves (4 along M x 2 along N). Block tile 128x128,
// wave tile 32x64. K staged in 128x64 LDS tiles, double-buffered, filled by
// global_load_async_to_lds_b128 (ASYNCcnt) while WMMAs consume the other
// buffer. 16 v_wmma per K-tile per wave; fragments come from ds_load_b128.
// ---------------------------------------------------------------------------
template <bool BIAS, bool RELU, bool RESID, bool BF16OUT>
__global__ __launch_bounds__(256, 1)
void gemm_nt(const bf16_t* __restrict__ A, const bf16_t* __restrict__ Bw,
             const float* __restrict__ bias, const float* __restrict__ resid,
             float* __restrict__ outF, bf16_t* __restrict__ outB,
             int M, int N, int K) {
  __shared__ bf16_t lA[2][128 * 64];   // 32 KB
  __shared__ bf16_t lB[2][128 * 64];   // 32 KB
  const int tid  = threadIdx.x;
  const int lane = tid & 31;
  const int wave = tid >> 5;
  const int wm = wave >> 1, wn = wave & 1;
  const int nb = N >> 7;
  const int bm = blockIdx.x / nb, bn = blockIdx.x % nb;
  const int nl = lane & 15, half = lane >> 4;
  (void)M;

  v8f acc[2][4];
#pragma unroll
  for (int i = 0; i < 2; ++i)
#pragma unroll
    for (int j = 0; j < 4; ++j) acc[i][j] = vzero8();

  const bf16_t* gA = A  + (size_t)(bm * 128) * K;
  const bf16_t* gB = Bw + (size_t)(bn * 128) * K;

  // Prime buffer 0 (8 async b128 per wave: 4 for A, 4 for B).
  stage_tile_async(lA[0], gA, K, tid);
  stage_tile_async(lB[0], gB, K, tid);

  const int nkt = K >> 6;                 // 64-wide K tiles
  for (int t = 0; t < nkt; ++t) {
    const int cur = t & 1;
    if (t + 1 < nkt) {
      stage_tile_async(lA[cur ^ 1], gA + (t + 1) * 64, K, tid);
      stage_tile_async(lB[cur ^ 1], gB + (t + 1) * 64, K, tid);
      ASYNC_WAIT_8();                     // oldest 8 (current buffer) landed
    } else {
      ASYNC_WAIT_0();
    }
    __syncthreads();                      // all waves' DMA portions visible

    const bf16_t* tA = lA[cur];
    const bf16_t* tB = lB[cur];
#pragma unroll
    for (int kk = 0; kk < 64; kk += 32) {
      const ABFrag a0 = load_afrag(tA + (wm * 32 +  0 + nl) * 64, lane, kk);
      const ABFrag a1 = load_afrag(tA + (wm * 32 + 16 + nl) * 64, lane, kk);
      const ABFrag b0 = load_bfrag(tB + (wn * 64 +  0 + nl) * 64, lane, kk);
      const ABFrag b1 = load_bfrag(tB + (wn * 64 + 16 + nl) * 64, lane, kk);
      const ABFrag b2 = load_bfrag(tB + (wn * 64 + 32 + nl) * 64, lane, kk);
      const ABFrag b3 = load_bfrag(tB + (wn * 64 + 48 + nl) * 64, lane, kk);
      acc[0][0] = wmma_bf16(a0, b0, acc[0][0]);
      acc[0][1] = wmma_bf16(a0, b1, acc[0][1]);
      acc[0][2] = wmma_bf16(a0, b2, acc[0][2]);
      acc[0][3] = wmma_bf16(a0, b3, acc[0][3]);
      acc[1][0] = wmma_bf16(a1, b0, acc[1][0]);
      acc[1][1] = wmma_bf16(a1, b1, acc[1][1]);
      acc[1][2] = wmma_bf16(a1, b2, acc[1][2]);
      acc[1][3] = wmma_bf16(a1, b3, acc[1][3]);
    }
    __syncthreads();                      // done reading before DMA overwrites
  }

  // Epilogue. C layout: lane holds column N=lane&15; VGPR p -> row p + 8*half.
#pragma unroll
  for (int i = 0; i < 2; ++i) {
#pragma unroll
    for (int j = 0; j < 4; ++j) {
      const int c = bn * 128 + wn * 64 + j * 16 + nl;
      float bia = 0.0f;
      if (BIAS) bia = bias[c];
#pragma unroll
      for (int p = 0; p < 8; ++p) {
        const int r = bm * 128 + wm * 32 + i * 16 + p + 8 * half;
        float v = acc[i][j][p] + bia;
        if (RELU) v = fmaxf(v, 0.0f);
        if (RESID) v += resid[(size_t)r * N + c];
        if (BF16OUT) outB[(size_t)r * N + c] = (bf16_t)v;
        else         outF[(size_t)r * N + c] = v;
      }
    }
  }
}

// ---------------------------------------------------------------------------
// v [B,S,H,64] bf16 -> vT [B,H,64,S] bf16 (64x64 tile transpose via LDS).
// ---------------------------------------------------------------------------
__global__ __launch_bounds__(256, 1)
void vtrans_kernel(const bf16_t* __restrict__ v, bf16_t* __restrict__ vT) {
  __shared__ bf16_t tile[64][72];
  const int st = blockIdx.x % (SEQ / 64);
  const int bh = blockIdx.x / (SEQ / 64);
  const int h = bh % NHEAD, b = bh / NHEAD;
  const int t = threadIdx.x;
  {
    const int sl = t >> 2;
    const int dq = (t & 3) * 16;
    const bf16_t* src = v + ((size_t)(b * SEQ + st * 64 + sl)) * D_MODEL + h * HEADD + dq;
    *(uint4*)&tile[sl][dq]     = *(const uint4*)src;
    *(uint4*)&tile[sl][dq + 8] = *(const uint4*)(src + 8);
  }
  __syncthreads();
  {
    const int dl = t >> 2;
    const int sq = (t & 3) * 16;
    union { bf16_t e[16]; uint4 u[2]; } ov;
#pragma unroll
    for (int i = 0; i < 16; ++i) ov.e[i] = tile[sq + i][dl];
    bf16_t* dst = vT + (((size_t)(b * NHEAD + h) * HEADD) + dl) * SEQ + st * 64 + sq;
    *(uint4*)dst       = ov.u[0];
    *(uint4*)(dst + 8) = ov.u[1];
  }
}

// ---------------------------------------------------------------------------
// Flash attention, wave32. Block = 128 thr = 4 waves; block owns 64 queries
// of one (b,h); each wave owns 16 queries. Scores computed TRANSPOSED
// (S^T = K * Q^T): the softmax'd tile is already in A-fragment layout for
// P*V -- register-only repack. SOFTWARE-PIPELINED: iteration i+1's K/V
// fragments are loaded into a second register set while iteration i's WMMAs
// and softmax run, so the pre-WMMA wait covers a full iteration of latency.
// ---------------------------------------------------------------------------
__global__ __launch_bounds__(128, 1)
void attn_kernel(const bf16_t* __restrict__ q, const bf16_t* __restrict__ k,
                 const bf16_t* __restrict__ vT, float* __restrict__ att) {
  const int lane = threadIdx.x & 31;
  const int wave = threadIdx.x >> 5;
  const int nl = lane & 15, half = lane >> 4;
  const int qt = blockIdx.x % (SEQ / 64);
  const int bh = blockIdx.x / (SEQ / 64);
  const int h = bh % NHEAD, b = bh / NHEAD;
  const int qbase = qt * 64 + wave * 16;
  const int myq = qbase + nl;

  const bf16_t* kbase_p = k  + (size_t)(b * SEQ) * D_MODEL + h * HEADD + (size_t)nl * D_MODEL;
  const bf16_t* vbase_p = vT + ((size_t)(b * NHEAD + h) * HEADD + nl) * SEQ;

  // Q^T as B-fragments (column = one query's head vector, K-contiguous).
  const bf16_t* qrow = q + ((size_t)(b * SEQ + qbase + nl)) * D_MODEL + h * HEADD;
  ABFrag bq[2];
  bq[0] = load_bfrag(qrow, lane, 0);
  bq[1] = load_bfrag(qrow, lane, 32);

  v8f o[4];
#pragma unroll
  for (int j = 0; j < 4; ++j) o[j] = vzero8();
  float m = -1e30f, lsum = 0.0f;

  const int kend = qbase + 16;                 // causal: keys <= qbase+15

  // ---- pipeline prologue: load fragments for k0 = 0 ----
  ABFrag ka[2][2], bv[4];
#pragma unroll
  for (int sub = 0; sub < 2; ++sub) {
    const bf16_t* krow = kbase_p + (size_t)(sub * 16) * D_MODEL;
    ka[sub][0] = load_afrag(krow, lane, 0);
    ka[sub][1] = load_afrag(krow, lane, 32);
  }
#pragma unroll
  for (int j = 0; j < 4; ++j) bv[j] = load_bfrag(vbase_p + (size_t)(j * 16) * SEQ, lane, 0);

  for (int k0 = 0; k0 < kend; k0 += 32) {
    // ---- prefetch next iteration's fragments into the shadow set ----
    const bool more = (k0 + 32) < kend;
    ABFrag nka[2][2], nbv[4];
    if (more) {
      const int kn = k0 + 32;
#pragma unroll
      for (int sub = 0; sub < 2; ++sub) {
        const bf16_t* krow = kbase_p + (size_t)(kn + sub * 16) * D_MODEL;
        nka[sub][0] = load_afrag(krow, lane, 0);
        nka[sub][1] = load_afrag(krow, lane, 32);
      }
#pragma unroll
      for (int j = 0; j < 4; ++j)
        nbv[j] = load_bfrag(vbase_p + (size_t)(j * 16) * SEQ, lane, kn);
    }

    // ---- S^T tiles: [32 keys x 16 queries], K = head dim 64 -> 4 WMMA ----
    v8f st[2];
#pragma unroll
    for (int sub = 0; sub < 2; ++sub) {
      v8f c = vzero8();
      c = wmma_bf16(ka[sub][0], bq[0], c);
      c = wmma_bf16(ka[sub][1], bq[1], c);
      st[sub] = c;
    }
    // scale + causal mask; per-lane column == one query
    float tmax = -1e30f;
    float sv[2][8];
#pragma unroll
    for (int sub = 0; sub < 2; ++sub)
#pragma unroll
      for (int p = 0; p < 8; ++p) {
        float s = st[sub][p] * 0.125f;          // 1/sqrt(64)
        const int key = k0 + sub * 16 + p + 8 * half;
        if (key > myq) s = -1e30f;
        sv[sub][p] = s;
        tmax = fmaxf(tmax, s);
      }
    tmax = fmaxf(tmax, __shfl_xor(tmax, 16, 32));
    const float mnew = fmaxf(m, tmax);
    const float corr = __expf(m - mnew);
    float rsum = 0.0f;
    ABFrag pf;                                   // P in A-fragment layout
#pragma unroll
    for (int sub = 0; sub < 2; ++sub)
#pragma unroll
      for (int p = 0; p < 8; ++p) {
        const float e = __expf(sv[sub][p] - mnew);
        rsum += e;
        pf.v[sub * 8 + p] = (bf16_t)e;           // exact A-layout element order
      }
    rsum += __shfl_xor(rsum, 16, 32);
    lsum = lsum * corr + rsum;
    m = mnew;
    // broadcast per-query-row rescale (row r's stats live in lane r)
    float cr[8];
#pragma unroll
    for (int p = 0; p < 8; ++p) cr[p] = __shfl(corr, p + 8 * half, 32);
#pragma unroll
    for (int j = 0; j < 4; ++j) {
#pragma unroll
      for (int p = 0; p < 8; ++p) o[j][p] *= cr[p];
      o[j] = wmma_bf16(pf, bv[j], o[j]);         // O += P * V
    }

    // ---- rotate shadow registers into the live set ----
    if (more) {
#pragma unroll
      for (int sub = 0; sub < 2; ++sub) {
        ka[sub][0] = nka[sub][0];
        ka[sub][1] = nka[sub][1];
      }
#pragma unroll
      for (int j = 0; j < 4; ++j) bv[j] = nbv[j];
    }
  }
  float lr[8];
#pragma unroll
  for (int p = 0; p < 8; ++p) lr[p] = 1.0f / __shfl(lsum, p + 8 * half, 32);
#pragma unroll
  for (int j = 0; j < 4; ++j)
#pragma unroll
    for (int p = 0; p < 8; ++p) {
      const int qr = qbase + p + 8 * half;
      att[((size_t)(b * SEQ + qr)) * D_MODEL + h * HEADD + j * 16 + nl] =
          o[j][p] * lr[p];
    }
}

// ---------------------------------------------------------------------------
// Launch pipeline (all on `stream`, graph-capture safe).
// ---------------------------------------------------------------------------
extern "C" void kernel_launch(void* const* d_in, const int* in_sizes, int n_in,
                              void* d_out, int out_size, void* d_ws, size_t ws_size,
                              hipStream_t stream) {
  (void)in_sizes; (void)n_in; (void)out_size; (void)ws_size;
  const float* x   = (const float*)d_in[0];
  const float* Wq  = (const float*)d_in[1];
  const float* Wk  = (const float*)d_in[2];
  const float* Wv  = (const float*)d_in[3];
  const float* g1  = (const float*)d_in[4];
  const float* be1 = (const float*)d_in[5];
  const float* g2  = (const float*)d_in[6];
  const float* be2 = (const float*)d_in[7];
  const float* W1  = (const float*)d_in[8];
  const float* bb1 = (const float*)d_in[9];
  const float* W2  = (const float*)d_in[10];
  const float* bb2 = (const float*)d_in[11];

  char* ws = (char*)d_ws;
  const size_t W_B   = (size_t)D_MODEL * D_MODEL * 2;   // 2 MB per weight (bf16)
  const size_t RD_BF = (size_t)ROWS * D_MODEL * 2;      // 16 MB
  const size_t RD_F  = (size_t)ROWS * D_MODEL * 4;      // 32 MB
  size_t off = 0;
  bf16_t* wq_bf = (bf16_t*)(ws + off); off += W_B;
  bf16_t* wk_bf = (bf16_t*)(ws + off); off += W_B;
  bf16_t* wv_bf = (bf16_t*)(ws + off); off += W_B;
  bf16_t* w1_bf = (bf16_t*)(ws + off); off += W_B;
  bf16_t* w2_bf = (bf16_t*)(ws + off); off += W_B;
  bf16_t* xn_bf = (bf16_t*)(ws + off); off += RD_BF;
  bf16_t* q_bf  = (bf16_t*)(ws + off); off += RD_BF;
  bf16_t* k_bf  = (bf16_t*)(ws + off); off += RD_BF;
  bf16_t* v_bf  = (bf16_t*)(ws + off); off += RD_BF;
  bf16_t* vT_bf = (bf16_t*)(ws + off); off += RD_BF;
  float*  att_f = (float*)(ws + off);  off += RD_F;
  float*  xt_f  = (float*)(ws + off);  off += RD_F;
  bf16_t* xt2_bf = xn_bf;   // reuse: xn dead after QKV
  bf16_t* h1_bf  = q_bf;    // reuse: q dead after attention

  const int nW = D_MODEL * D_MODEL;
  cvt_bf16_kernel<<<nW / 1024, 256, 0, stream>>>(Wq, wq_bf, nW);
  cvt_bf16_kernel<<<nW / 1024, 256, 0, stream>>>(Wk, wk_bf, nW);
  cvt_bf16_kernel<<<nW / 1024, 256, 0, stream>>>(Wv, wv_bf, nW);
  cvt_bf16_kernel<<<nW / 1024, 256, 0, stream>>>(W1, w1_bf, nW);
  cvt_bf16_kernel<<<nW / 1024, 256, 0, stream>>>(W2, w2_bf, nW);

  ln_bf16_kernel<<<ROWS, 256, 0, stream>>>(x, g1, be1, xn_bf);

  const int gGemm = (ROWS / 128) * (D_MODEL / 128);     // 64 * 8 = 512
  gemm_nt<false, false, false, true><<<gGemm, 256, 0, stream>>>(
      xn_bf, wq_bf, nullptr, nullptr, nullptr, q_bf, ROWS, D_MODEL, D_MODEL);
  gemm_nt<false, false, false, true><<<gGemm, 256, 0, stream>>>(
      xn_bf, wk_bf, nullptr, nullptr, nullptr, k_bf, ROWS, D_MODEL, D_MODEL);
  gemm_nt<false, false, false, true><<<gGemm, 256, 0, stream>>>(
      xn_bf, wv_bf, nullptr, nullptr, nullptr, v_bf, ROWS, D_MODEL, D_MODEL);

  const int gBH = BATCH * NHEAD * (SEQ / 64);           // 2048
  vtrans_kernel<<<gBH, 256, 0, stream>>>(v_bf, vT_bf);
  attn_kernel<<<gBH, 128, 0, stream>>>(q_bf, k_bf, vT_bf, att_f);

  add_ln_kernel<<<ROWS, 256, 0, stream>>>(x, att_f, g2, be2, xt_f, xt2_bf);

  gemm_nt<true, true, false, true><<<gGemm, 256, 0, stream>>>(
      xt2_bf, w1_bf, bb1, nullptr, nullptr, h1_bf, ROWS, D_MODEL, D_MODEL);
  gemm_nt<true, false, true, false><<<gGemm, 256, 0, stream>>>(
      h1_bf, w2_bf, bb2, xt_f, (float*)d_out, nullptr, ROWS, D_MODEL, D_MODEL);
}
